// RAPMemory_34256659153442
// MI455X (gfx1250) — matrix operand order repaired and checked by reading, not compiled, not run on hardware
//
#include <hip/hip_runtime.h>
#include <hip/hip_bf16.h>

// ---------------------------------------------------------------------------
// MI455X (gfx1250) implementation.
//  - LTC scan: VALU/transcendental bound (sigmoid inside reduction -> no WMMA).
//    One block per batch element, 512 threads (unit j), v-state in LDS,
//    weights stream from L2 (8MB << 192MB L2), column-coalesced loads.
//  - Hopfield attention + MLP: f16 WMMA (v_wmma_f32_16x16x32_f16) GEMMs with
//    ISA-documented fragment layouts (CDNA5 ISA §7.12.2).
// ---------------------------------------------------------------------------

typedef _Float16 v16h __attribute__((ext_vector_type(16)));
typedef float    v8f  __attribute__((ext_vector_type(8)));

#define LTC_U   512
#define LTC_T   64
#define LTC_HID 256
#define GEMM_K  256
#define GEMM_ROWS 64
#define M_TOTAL 2048   // B*T

__device__ __forceinline__ float sigmoidf_(float x) {
    return 1.0f / (1.0f + __expf(-x));
}

// ---------------------------------------------------------------------------
// Kernel 1: LTC implicit-Euler scan.  grid = B (32), block = 512 threads.
// ---------------------------------------------------------------------------
__global__ __launch_bounds__(512)
void ltc_kernel(const float* __restrict__ x,
                const float* __restrict__ input_w, const float* __restrict__ input_b,
                const float* __restrict__ s_mu,    const float* __restrict__ s_sigma,
                const float* __restrict__ s_w,     const float* __restrict__ s_erev,
                const float* __restrict__ r_mu,    const float* __restrict__ r_sigma,
                const float* __restrict__ r_w,     const float* __restrict__ r_erev,
                const float* __restrict__ gleak,   const float* __restrict__ vleak,
                const float* __restrict__ cm,
                const float* __restrict__ out_w,   const float* __restrict__ out_b,
                float* __restrict__ ltc_out,       float* __restrict__ hidden)
{
    __shared__ float vsh[LTC_U];
    __shared__ float inp[LTC_U];
    const int b = blockIdx.x;
    const int j = threadIdx.x;

    const float gl   = gleak[j];
    const float cmt  = cm[j] * 6.0f;          // cm * ODE_UNFOLDS
    const float glvl = gl * vleak[j];
    const float ow   = (j < LTC_HID) ? out_w[j] : 0.0f;
    const float ob   = (j < LTC_HID) ? out_b[j] : 0.0f;

    vsh[j] = 0.0f;
    __syncthreads();

    for (int t = 0; t < LTC_T; ++t) {
        const float xin = x[((size_t)b * LTC_T + t) * LTC_U + j];
        if (t + 1 < LTC_T)
            __builtin_prefetch(&x[((size_t)b * LTC_T + t + 1) * LTC_U + j], 0, 0);
        inp[j] = xin * input_w[j] + input_b[j];
        __syncthreads();

        // sensory synapses: reduce over presynaptic i, column j coalesced.
        float wnum = 0.0f, wden = 0.0f;
        #pragma unroll 4
        for (int i = 0; i < LTC_U; ++i) {
            const size_t idx = (size_t)i * LTC_U + j;
            const float sa = s_w[idx] * sigmoidf_(s_sigma[idx] * (inp[i] - s_mu[idx]));
            wnum += sa * s_erev[idx];
            wden += sa;
        }

        #pragma unroll 1
        for (int u = 0; u < 6; ++u) {
            float num = cmt * vsh[j] + glvl + wnum;
            float den = cmt + gl + wden;
            #pragma unroll 4
            for (int i = 0; i < LTC_U; ++i) {
                const size_t idx = (size_t)i * LTC_U + j;
                const float wa = r_w[idx] * sigmoidf_(r_sigma[idx] * (vsh[i] - r_mu[idx]));
                num += wa * r_erev[idx];
                den += wa;
            }
            const float vn = num / (den + 1e-8f);
            __syncthreads();
            vsh[j] = vn;
            __syncthreads();
        }

        if (j < LTC_HID)
            ltc_out[((size_t)b * LTC_T + t) * LTC_HID + j] = vsh[j] * ow + ob;
        __syncthreads();
    }
    hidden[(size_t)b * LTC_U + j] = vsh[j];
}

// ---------------------------------------------------------------------------
// Kernel 2: f32 -> f16 weight conversion (once per launch, into workspace).
// ---------------------------------------------------------------------------
__global__ void f32_to_f16_kernel(const float* __restrict__ src,
                                  _Float16* __restrict__ dst, int n)
{
    const int i = blockIdx.x * 256 + threadIdx.x;
    if (i < n) dst[i] = (_Float16)src[i];
}

// ---------------------------------------------------------------------------
// Kernel 3: generic fused GEMM:  out[M,N] = act( LN?(in)[M,256] @ Wh[256,N]
//                                               + bias ) (+ residual)
// WMMA f32_16x16x32_f16; block = 256 threads (8 waves), 64 rows per block.
// Fragment layouts per CDNA5 ISA §7.12.2:
//   A: lane L -> m=L&15, h=L>>4; vgpr j -> k = (j>>2)*16 + h*8 + (j&3)*2 (+kk*32)
//   B: lane L -> n = L&15, h=L>>4; vgpr j -> k as for A (pairs k, k+1)
//   C: lane L -> n = L&15;       vgpr j -> m = j + 8*(L>>4)
// ---------------------------------------------------------------------------
template<int N, bool LN, bool RES, bool SILU>
__global__ __launch_bounds__(256)
void gemm_wmma(const float* __restrict__ inp, const _Float16* __restrict__ Wh,
               const float* __restrict__ bias,
               const float* __restrict__ lnG, const float* __restrict__ lnB,
               const float* __restrict__ resid, float* __restrict__ out)
{
    __shared__ _Float16 As[GEMM_ROWS][GEMM_K];
    const int tid  = threadIdx.x;
    const int row0 = blockIdx.x * GEMM_ROWS;

    if (LN) {
        // 4 threads per row: partial sums + quad shuffle reduce, then write f16.
        const int r  = tid >> 2;
        const int q4 = tid & 3;
        const float* rp = inp + (size_t)(row0 + r) * GEMM_K;
        float s = 0.0f, ss = 0.0f;
        for (int kk = 0; kk < GEMM_K / 4; ++kk) {
            const float xv = rp[q4 + 4 * kk];
            s += xv; ss += xv * xv;
        }
        s  += __shfl_xor(s, 1);  s  += __shfl_xor(s, 2);
        ss += __shfl_xor(ss, 1); ss += __shfl_xor(ss, 2);
        const float mean = s * (1.0f / GEMM_K);
        const float var  = ss * (1.0f / GEMM_K) - mean * mean;
        const float inv  = rsqrtf(var + 1e-5f);
        for (int kk = 0; kk < GEMM_K / 4; ++kk) {
            const int k = q4 + 4 * kk;
            const float xv = rp[k];
            As[r][k] = (_Float16)(((xv - mean) * inv) * lnG[k] + lnB[k]);
        }
    } else {
        for (int idx = tid; idx < GEMM_ROWS * GEMM_K; idx += 256) {
            const int r = idx >> 8, k = idx & 255;
            As[r][k] = (_Float16)inp[(size_t)(row0 + r) * GEMM_K + k];
        }
    }
    __syncthreads();

    const int wave = tid >> 5;
    const int lane = tid & 31;
    const int m16  = lane & 15;
    const int hi   = lane >> 4;
    constexpr int NT = N / 16;

    for (int tile = wave; tile < 4 * NT; tile += 8) {
        const int mt = tile / NT, nt = tile % NT;
        v8f acc = {};
        #pragma unroll
        for (int kk = 0; kk < GEMM_K / 32; ++kk) {
            v16h a, bfr;
            #pragma unroll
            for (int jj = 0; jj < 8; ++jj) {
                const int kidx = kk * 32 + ((jj >> 2) << 4) + (hi << 3) + ((jj & 3) << 1);
                a[2 * jj]     = As[mt * 16 + m16][kidx];
                a[2 * jj + 1] = As[mt * 16 + m16][kidx + 1];
            }
            const _Float16* wrow = Wh + nt * 16 + m16;
            #pragma unroll
            for (int jj = 0; jj < 8; ++jj) {
                const int kidx = kk * 32 + ((jj >> 2) << 4) + (hi << 3) + ((jj & 3) << 1);
                bfr[2 * jj]     = wrow[(size_t)kidx * N];
                bfr[2 * jj + 1] = wrow[(size_t)(kidx + 1) * N];
            }
            acc = __builtin_amdgcn_wmma_f32_16x16x32_f16(false, a, false, bfr,
                                                         (short)0, acc, false, false);
        }
        const int n  = nt * 16 + m16;
        const float bn = bias[n];
        #pragma unroll
        for (int jj = 0; jj < 8; ++jj) {
            const int m = mt * 16 + (hi << 3) + jj;
            const size_t oidx = (size_t)(row0 + m) * N + n;
            float val = acc[jj] + bn;
            if (RES)  val += resid[oidx];
            if (SILU) val = val * (1.0f / (1.0f + __expf(-val)));
            out[oidx] = val;
        }
    }
}

// ---------------------------------------------------------------------------
// Kernel 4: per-(batch,head) 64x64 self-attention with WMMA. grid = B*HEADS,
// block = 128 threads (4 waves). q pre-scaled by beta at staging.
// ---------------------------------------------------------------------------
__global__ __launch_bounds__(128)
void attn_kernel(const float* __restrict__ q, const float* __restrict__ k,
                 const float* __restrict__ v, float* __restrict__ o)
{
    const int bh = blockIdx.x;
    const int b  = bh >> 2;
    const int h  = bh & 3;
    __shared__ _Float16 qs[64][64];
    __shared__ _Float16 ks[64][64];
    __shared__ _Float16 vs[64][64];
    __shared__ float    sc[64][64];
    __shared__ _Float16 aw[64][64];

    const int tid = threadIdx.x;
    const float beta = 0.125f;   // 1/sqrt(64)

    for (int idx = tid; idx < 4096; idx += 128) {
        const int t = idx >> 6, d = idx & 63;
        const size_t g = ((size_t)(b * 64 + t)) * 256 + h * 64 + d;
        qs[t][d] = (_Float16)(q[g] * beta);
        ks[t][d] = (_Float16)k[g];
        vs[t][d] = (_Float16)v[g];
    }
    __syncthreads();

    const int wave = tid >> 5;
    const int lane = tid & 31;
    const int m16  = lane & 15;
    const int hi   = lane >> 4;

    // scores = (q*beta) @ k^T
    for (int tile = wave; tile < 16; tile += 4) {
        const int mt = tile >> 2, nt = tile & 3;
        v8f acc = {};
        #pragma unroll
        for (int kk = 0; kk < 2; ++kk) {
            v16h a, bfr;
            #pragma unroll
            for (int jj = 0; jj < 8; ++jj) {
                const int kidx = kk * 32 + ((jj >> 2) << 4) + (hi << 3) + ((jj & 3) << 1);
                a[2 * jj]     = qs[mt * 16 + m16][kidx];
                a[2 * jj + 1] = qs[mt * 16 + m16][kidx + 1];
            }
            #pragma unroll
            for (int jj = 0; jj < 8; ++jj) {   // B[d][t'] = ks[t'][d]
                const int kidx = kk * 32 + ((jj >> 2) << 4) + (hi << 3) + ((jj & 3) << 1);
                bfr[2 * jj]     = ks[nt * 16 + m16][kidx];
                bfr[2 * jj + 1] = ks[nt * 16 + m16][kidx + 1];
            }
            acc = __builtin_amdgcn_wmma_f32_16x16x32_f16(false, a, false, bfr,
                                                         (short)0, acc, false, false);
        }
        #pragma unroll
        for (int jj = 0; jj < 8; ++jj)
            sc[mt * 16 + (hi << 3) + jj][nt * 16 + m16] = acc[jj];
    }
    __syncthreads();

    // row softmax (64 rows, one per thread 0..63)
    if (tid < 64) {
        float mx = -1e30f;
        for (int c = 0; c < 64; ++c) mx = fmaxf(mx, sc[tid][c]);
        float s = 0.0f;
        for (int c = 0; c < 64; ++c) {
            const float e = __expf(sc[tid][c] - mx);
            sc[tid][c] = e;
            s += e;
        }
        const float invs = 1.0f / s;
        for (int c = 0; c < 64; ++c)
            aw[tid][c] = (_Float16)(sc[tid][c] * invs);
    }
    __syncthreads();

    // out = attn @ v
    for (int tile = wave; tile < 16; tile += 4) {
        const int mt = tile >> 2, nt = tile & 3;
        v8f acc = {};
        #pragma unroll
        for (int kk = 0; kk < 2; ++kk) {
            v16h a, bfr;
            #pragma unroll
            for (int jj = 0; jj < 8; ++jj) {
                const int kidx = kk * 32 + ((jj >> 2) << 4) + (hi << 3) + ((jj & 3) << 1);
                a[2 * jj]     = aw[mt * 16 + m16][kidx];
                a[2 * jj + 1] = aw[mt * 16 + m16][kidx + 1];
            }
            #pragma unroll
            for (int jj = 0; jj < 8; ++jj) {   // B[t'][d] = vs[t'][d]
                const int kidx = kk * 32 + ((jj >> 2) << 4) + (hi << 3) + ((jj & 3) << 1);
                bfr[2 * jj]     = vs[kidx][nt * 16 + m16];
                bfr[2 * jj + 1] = vs[kidx + 1][nt * 16 + m16];
            }
            acc = __builtin_amdgcn_wmma_f32_16x16x32_f16(false, a, false, bfr,
                                                         (short)0, acc, false, false);
        }
        #pragma unroll
        for (int jj = 0; jj < 8; ++jj) {
            const int m = mt * 16 + (hi << 3) + jj;
            o[((size_t)(b * 64 + m)) * 256 + h * 64 + nt * 16 + m16] = acc[jj];
        }
    }
}

// ---------------------------------------------------------------------------
// Host orchestration.
// ---------------------------------------------------------------------------
extern "C" void kernel_launch(void* const* d_in, const int* in_sizes, int n_in,
                              void* d_out, int out_size, void* d_ws, size_t ws_size,
                              hipStream_t stream)
{
    (void)in_sizes; (void)n_in; (void)out_size; (void)ws_size;

    const float* x        = (const float*)d_in[0];
    const float* input_w  = (const float*)d_in[1];
    const float* input_b  = (const float*)d_in[2];
    const float* s_mu     = (const float*)d_in[3];
    const float* s_sigma  = (const float*)d_in[4];
    const float* s_w      = (const float*)d_in[5];
    const float* s_erev   = (const float*)d_in[6];
    const float* r_mu     = (const float*)d_in[7];
    const float* r_sigma  = (const float*)d_in[8];
    const float* r_w      = (const float*)d_in[9];
    const float* r_erev   = (const float*)d_in[10];
    const float* gleak    = (const float*)d_in[11];
    const float* vleak    = (const float*)d_in[12];
    const float* cm       = (const float*)d_in[13];
    const float* out_w    = (const float*)d_in[14];
    const float* out_b    = (const float*)d_in[15];
    const float* ln_q_g   = (const float*)d_in[16];
    const float* ln_q_b   = (const float*)d_in[17];
    const float* ln_k_g   = (const float*)d_in[18];
    const float* ln_k_b   = (const float*)d_in[19];
    const float* ln_v_g   = (const float*)d_in[20];
    const float* ln_v_b   = (const float*)d_in[21];
    const float* wq       = (const float*)d_in[22];
    const float* bq       = (const float*)d_in[23];
    const float* wk       = (const float*)d_in[24];
    const float* bk       = (const float*)d_in[25];
    const float* wv       = (const float*)d_in[26];
    const float* bv       = (const float*)d_in[27];
    const float* wo       = (const float*)d_in[28];
    const float* bo       = (const float*)d_in[29];
    const float* w1       = (const float*)d_in[30];
    const float* b1       = (const float*)d_in[31];
    const float* w2       = (const float*)d_in[32];
    const float* b2       = (const float*)d_in[33];

    float* out_combined = (float*)d_out;                 // 2048*256
    float* out_belief   = out_combined + 524288;         // 2048*64
    float* out_hidden   = out_belief + 131072;           // 32*512

    char* ws = (char*)d_ws;
    float* ltc_out = (float*)(ws);                        // 2048*256 f32
    float* qb      = (float*)(ws + 2097152);
    float* kb      = (float*)(ws + 4194304);
    float* vb      = (float*)(ws + 6291456);
    float* ob      = (float*)(ws + 8388608);
    _Float16* wqh  = (_Float16*)(ws + 10485760);
    _Float16* wkh  = wqh + 65536;
    _Float16* wvh  = wkh + 65536;
    _Float16* woh  = wvh + 65536;
    _Float16* w1h  = woh + 65536;
    _Float16* w2h  = w1h + 65536;
    float* h1      = qb;  // reuse q buffer for MLP hidden

    // --- weight conversions (f32 -> f16), deterministic every call ---
    f32_to_f16_kernel<<<256, 256, 0, stream>>>(wq, wqh, 65536);
    f32_to_f16_kernel<<<256, 256, 0, stream>>>(wk, wkh, 65536);
    f32_to_f16_kernel<<<256, 256, 0, stream>>>(wv, wvh, 65536);
    f32_to_f16_kernel<<<256, 256, 0, stream>>>(wo, woh, 65536);
    f32_to_f16_kernel<<<256, 256, 0, stream>>>(w1, w1h, 65536);
    f32_to_f16_kernel<<<64,  256, 0, stream>>>(w2, w2h, 16384);

    // --- LTC scan ---
    ltc_kernel<<<32, 512, 0, stream>>>(x, input_w, input_b,
                                       s_mu, s_sigma, s_w, s_erev,
                                       r_mu, r_sigma, r_w, r_erev,
                                       gleak, vleak, cm, out_w, out_b,
                                       ltc_out, out_hidden);

    // --- Hopfield: LN + projections (WMMA) ---
    gemm_wmma<256, true,  false, false><<<32, 256, 0, stream>>>(
        ltc_out, wqh, bq, ln_q_g, ln_q_b, nullptr, qb);
    gemm_wmma<256, true,  false, false><<<32, 256, 0, stream>>>(
        ltc_out, wkh, bk, ln_k_g, ln_k_b, nullptr, kb);
    gemm_wmma<256, true,  false, false><<<32, 256, 0, stream>>>(
        ltc_out, wvh, bv, ln_v_g, ln_v_b, nullptr, vb);

    // --- attention ---
    attn_kernel<<<128, 128, 0, stream>>>(qb, kb, vb, ob);

    // --- output projection + residual: combined = o@wo + bo + ltc_out ---
    gemm_wmma<256, false, true,  false><<<32, 256, 0, stream>>>(
        ob, woh, bo, nullptr, nullptr, ltc_out, out_combined);

    // --- belief MLP: h1 = silu(combined@w1+b1); belief = h1@w2+b2 ---
    gemm_wmma<256, false, false, true ><<<32, 256, 0, stream>>>(
        out_combined, w1h, b1, nullptr, nullptr, nullptr, h1);
    gemm_wmma<64,  false, false, false><<<32, 256, 0, stream>>>(
        h1, w2h, b2, nullptr, nullptr, nullptr, out_belief);
}
